// CWICMLP_19138374271286
// MI455X (gfx1250) — compile-verified, matrix operand order checked by reading, not run on hardware
//
#include <hip/hip_runtime.h>
#include <hip/hip_bf16.h>
#include <math.h>

typedef __attribute__((ext_vector_type(16))) __bf16 v16bf;
typedef __attribute__((ext_vector_type(2)))  __bf16 v2bf;
typedef __attribute__((ext_vector_type(8)))  float  v8f;
typedef __attribute__((ext_vector_type(4)))  unsigned int u32x4;
typedef __attribute__((ext_vector_type(8)))  int i32x8;
typedef __attribute__((ext_vector_type(4)))  int i32x4;

#define T_TOK   2048
#define IN_F    2048
#define INTER_F 8192
#define OUT_F   2048
#define STRIPE_W 256
#define N_STRIPES 32

#define MT 128            // token tile
#define NT 128            // column tile
#define KB 32             // K per WMMA step
#define TSTR 40           // LDS row stride in bf16: 80B = 64B data + 16B pad (16B-aligned)

#if defined(__HIP_DEVICE_COMPILE__) && __has_builtin(__builtin_amdgcn_tensor_load_to_lds) && __has_builtin(__builtin_amdgcn_s_wait_tensorcnt)
#define USE_TDM 1
#else
#define USE_TDM 0
#endif

__device__ __forceinline__ unsigned short f2bfu(float f) {
  unsigned u = __builtin_bit_cast(unsigned, f);
  unsigned r = u + 0x7fffu + ((u >> 16) & 1u);   // round-to-nearest-even
  return (unsigned short)(r >> 16);
}

#if defined(__HIP_DEVICE_COMPILE__) && __has_builtin(__builtin_amdgcn_cvt_pk_bf16_f32)
__device__ __forceinline__ unsigned pack_bf16(float a, float b) {
  v2bf v = __builtin_amdgcn_cvt_pk_bf16_f32(a, b);   // v_cvt_pk_bf16_f32
  return __builtin_bit_cast(unsigned, v);
}
#else
__device__ __forceinline__ unsigned pack_bf16(float a, float b) {
  return (unsigned)f2bfu(a) | ((unsigned)f2bfu(b) << 16);
}
#endif

// Per-lane WMMA 16-bit operand fragment: two ds_load_b128 from a padded row.
struct FragBits { u32x4 lo; u32x4 hi; };
__device__ __forceinline__ v16bf load_frag16(const unsigned short* rowbase, int hf) {
  const u32x4* q = (const u32x4*)rowbase;     // row = 80B: [0,64) data, [64,80) pad
  FragBits s;
  s.lo = q[hf];        // k = 8*hf    .. 8*hf+7
  s.hi = q[2 + hf];    // k = 16+8*hf .. 16+8*hf+7
  return __builtin_bit_cast(v16bf, s);
}

#if USE_TDM
// 2D TDM load: tile_rows rows of tile_k bf16 elements, global row stride
// row_stride elements, into LDS with 16B pad after each 64B row.
__device__ __forceinline__ void tdm_load_2d(const void* gaddr, unsigned lds_byte_addr,
                                            unsigned tile_k, unsigned tile_rows,
                                            unsigned row_stride_elems) {
  unsigned long long ga = (unsigned long long)(uintptr_t)gaddr;
  u32x4 g0;
  g0[0] = 1u;                                        // count=1 (valid user D#)
  g0[1] = lds_byte_addr;                             // lds_addr
  g0[2] = (unsigned)(ga & 0xFFFFFFFFu);              // global_addr[31:0]
  g0[3] = (unsigned)((ga >> 32) & 0x01FFFFFFu) | (2u << 30);  // addr[56:32], type=2
  const unsigned td0 = 1u << 20, td1 = 1u << 20;     // generous tensor dims (no OOB)
  i32x8 g1;
  g1[0] = (int)((1u << 16)                           // data_size = 2 bytes
              | (1u << 20)                           // pad_enable
              | (3u << 22)                           // pad_interval: every 16 DWORDs (64B)
              | (3u << 25));                         // pad_amount: 4 DWORDs (16B)
  g1[1] = (int)((td0 & 0xFFFFu) << 16);
  g1[2] = (int)(((td0 >> 16) & 0xFFFFu) | ((td1 & 0xFFFFu) << 16));
  g1[3] = (int)(((td1 >> 16) & 0xFFFFu) | ((tile_k & 0xFFFFu) << 16));   // tile_dim0
  g1[4] = (int)(tile_rows & 0xFFFFu);                // tile_dim1 (tile_dim2=0)
  g1[5] = (int)row_stride_elems;                     // tensor_dim0_stride[31:0]
  g1[6] = 0;
  g1[7] = 0;
  i32x4 z4 = {0, 0, 0, 0};
  i32x8 z8 = {0, 0, 0, 0, 0, 0, 0, 0};
  __builtin_amdgcn_tensor_load_to_lds(g0, g1, z4, z4, z8, 0);
}
#endif

// ---------------------------------------------------------------------------
// Kernel 0: init side outputs + precompute mu / thr tables
// ---------------------------------------------------------------------------
__global__ void cwic_init_kernel(float* __restrict__ dense, float* __restrict__ active) {
  int i = blockIdx.x * blockDim.x + threadIdx.x;
  if (i < T_TOK) {
    dense[i]  = 16777216.0f;   // STRIPE * N_STRIPES * IN_F
    active[i] = 0.0f;
  }
}

__global__ void cwic_tab_kernel(const float* __restrict__ gthr, const float* __restrict__ med,
                                const float* __restrict__ aad,
                                float* __restrict__ mu_tab, float* __restrict__ thr_tab,
                                float debiaser, float inv_s2pi, float thr_scale) {
  int i = blockIdx.x * blockDim.x + threadIdx.x;         // over N_STRIPES * IN_F
  if (i < N_STRIPES * IN_F) {
    int gi = i & (IN_F - 1);
    if (i < IN_F) mu_tab[i] = med[i] * debiaser;
    thr_tab[i] = gthr[i] * (aad[gi] * debiaser * inv_s2pi) * thr_scale;
  }
}

// ---------------------------------------------------------------------------
// Transpose + convert: src f32 [K][N] row-major -> dst bf16 [N][K]
// ---------------------------------------------------------------------------
__global__ __launch_bounds__(256) void cwic_transpose_bf16_kernel(
    const float* __restrict__ src, unsigned short* __restrict__ dst, int K, int N) {
  __shared__ float t[32][33];
  const int n0 = blockIdx.x * 32, k0 = blockIdx.y * 32;
  const int tx = threadIdx.x & 31, ty = threadIdx.x >> 5;
  #pragma unroll
  for (int i = ty; i < 32; i += 8)
    t[i][tx] = src[(size_t)(k0 + i) * N + n0 + tx];
  __syncthreads();
  #pragma unroll
  for (int i = ty; i < 32; i += 8)
    dst[(size_t)(n0 + i) * K + k0 + tx] = f2bfu(t[tx][i]);
}

// Straight convert f32 -> bf16 (pairs)
__global__ void cwic_convert_bf16_kernel(const float* __restrict__ src,
                                         unsigned short* __restrict__ dst, int npairs) {
  int i = blockIdx.x * blockDim.x + threadIdx.x;
  if (i < npairs) {
    float2 v = ((const float2*)src)[i];
    ((unsigned*)dst)[i] = pack_bf16(v.x, v.y);
  }
}

// ---------------------------------------------------------------------------
// Kernel 1: fused gate (stripe-masked) + up GEMMs, h = silu(g)*u -> bf16 ws
// 256 threads (8 waves), tile 128 tokens x 128 cols.
// B tiles: double-buffered TDM pipeline overlapping WMMA.
// ---------------------------------------------------------------------------
__global__ __launch_bounds__(256) void cwic_gate_up_kernel(
    const float* __restrict__ x,
    const unsigned short* __restrict__ gwt,   // gate W^T bf16 [INTER_F][IN_F]
    const unsigned short* __restrict__ uwt,   // up   W^T bf16 [INTER_F][IN_F]
    const float* __restrict__ gb,  const float* __restrict__ ub,
    const float* __restrict__ mu_tab, const float* __restrict__ thr_tab,
    unsigned short* __restrict__ h_out, float* __restrict__ active)
{
  __shared__ __align__(16) unsigned short As_m[MT * TSTR];
  __shared__ __align__(16) unsigned short As_p[MT * TSTR];
  __shared__ __align__(16) unsigned short Bg[2][NT * TSTR];   // [n][k], double-buffered
  __shared__ __align__(16) unsigned short Bu[2][NT * TSTR];
  __shared__ float cnt[MT];

  const int tid = threadIdx.x;
  const int c0  = blockIdx.x * NT;
  const int t0  = blockIdx.y * MT;
  const int stripe = c0 / STRIPE_W;
  const bool count_active = ((blockIdx.x & 1) == 0);

  for (int i = tid; i < MT; i += 256) cnt[i] = 0.0f;

  const int wid = tid >> 5, lane = tid & 31;
  const int hf = lane >> 4, r = lane & 15;
  const int wm = wid & 3,  wn = wid >> 2;        // 4x2 waves: 32 tok x 64 col each

  v8f accg[2][4], accu[2][4];
  #pragma unroll
  for (int i = 0; i < 2; ++i)
    #pragma unroll
    for (int j = 0; j < 4; ++j) { v8f z = {}; accg[i][j] = z; accu[i][j] = z; }

  float mcnt[8];
  #pragma unroll
  for (int m = 0; m < 8; ++m) mcnt[m] = 0.0f;
  const int arow0 = tid >> 4;        // this thread's base row (+16*m)
  const int cp    = tid & 15;        // this thread's k-pair within a row

#if USE_TDM
  if (tid == 0) {   // prologue: DMA first weight tiles into buffer 0
    tdm_load_2d(&gwt[(size_t)c0 * IN_F], (unsigned)(size_t)(void*)Bg[0], KB, NT, IN_F);
    tdm_load_2d(&uwt[(size_t)c0 * IN_F], (unsigned)(size_t)(void*)Bu[0], KB, NT, IN_F);
  }
#endif

  int p = 0;
  for (int k0 = 0; k0 < IN_F; k0 += KB, p ^= 1) {
    __syncthreads();   // all waves done with buffer p^1 (read last iter) and A tiles
#if USE_TDM
    if (tid == 0 && k0 + KB < IN_F) {   // DMA next tiles while we mask + multiply
      tdm_load_2d(&gwt[(size_t)c0 * IN_F + k0 + KB], (unsigned)(size_t)(void*)Bg[p ^ 1], KB, NT, IN_F);
      tdm_load_2d(&uwt[(size_t)c0 * IN_F + k0 + KB], (unsigned)(size_t)(void*)Bu[p ^ 1], KB, NT, IN_F);
    }
#else
    for (int idx = tid; idx < NT * (KB / 8); idx += 256) {
      int nn = idx >> 2, ch = idx & 3;
      *(u32x4*)&Bg[p][nn * TSTR + ch * 8] = *(const u32x4*)&gwt[(size_t)(c0 + nn) * IN_F + k0 + ch * 8];
      *(u32x4*)&Bu[p][nn * TSTR + ch * 8] = *(const u32x4*)&uwt[(size_t)(c0 + nn) * IN_F + k0 + ch * 8];
    }
#endif
    // ---- stage A tiles: mask with precomputed tables, pack 2 bf16 / store ----
    #pragma unroll
    for (int m = 0; m < 8; ++m) {
      int row = arow0 + 16 * m;
      int gi  = k0 + cp * 2;
      float2 xv = *(const float2*)&x[(size_t)(t0 + row) * IN_F + gi];
      float2 mu = *(const float2*)&mu_tab[gi];
      float2 th = *(const float2*)&thr_tab[(size_t)stripe * IN_F + gi];
      bool m0 = fabsf(xv.x - mu.x) > th.x;
      bool m1 = fabsf(xv.y - mu.y) > th.y;
      *(unsigned*)&As_m[row * TSTR + cp * 2] = pack_bf16(m0 ? xv.x : mu.x, m1 ? xv.y : mu.y);
      *(unsigned*)&As_p[row * TSTR + cp * 2] = pack_bf16(xv.x, xv.y);
      mcnt[m] += (m0 ? 1.0f : 0.0f) + (m1 ? 1.0f : 0.0f);
    }
    if (k0 + KB < IN_F)
      __builtin_prefetch(&x[(size_t)t0 * IN_F + k0 + KB], 0, 1);  // global_prefetch_b8
#if USE_TDM
    if (tid == 0) {
      if (k0 + KB < IN_F) __builtin_amdgcn_s_wait_tensorcnt(2);  // this tile done; next in flight
      else                __builtin_amdgcn_s_wait_tensorcnt(0);
    }
#endif
    __syncthreads();

    // ---- fragments: 2x ds_load_b128 each ----
    v16bf am[2], ap[2];
    #pragma unroll
    for (int i = 0; i < 2; ++i) {
      int row = wm * 32 + i * 16 + r;
      am[i] = load_frag16(&As_m[row * TSTR], hf);
      ap[i] = load_frag16(&As_p[row * TSTR], hf);
    }
    #pragma unroll
    for (int j = 0; j < 4; ++j) {
      int nn = wn * 64 + j * 16 + r;
      v16bf bg = load_frag16(&Bg[p][nn * TSTR], hf);
      v16bf bu = load_frag16(&Bu[p][nn * TSTR], hf);
      #pragma unroll
      for (int i = 0; i < 2; ++i) {
        accg[i][j] = __builtin_amdgcn_wmma_f32_16x16x32_bf16(
            false, am[i], false, bg, (short)0, accg[i][j], false, false);
        accu[i][j] = __builtin_amdgcn_wmma_f32_16x16x32_bf16(
            false, ap[i], false, bu, (short)0, accu[i][j], false, false);
      }
    }
  }

  // ---- epilogue: h = silu(g+gb) * (u+ub) ----
  #pragma unroll
  for (int i = 0; i < 2; ++i) {
    #pragma unroll
    for (int j = 0; j < 4; ++j) {
      int col = c0 + wn * 64 + j * 16 + r;
      float gbv = gb[col], ubv = ub[col];
      #pragma unroll
      for (int e = 0; e < 8; ++e) {
        int tok = t0 + wm * 32 + i * 16 + e + 8 * hf;
        float g = accg[i][j][e] + gbv;
        float u = accu[i][j][e] + ubv;
        float s = g / (1.0f + __expf(-g));
        h_out[(size_t)tok * INTER_F + col] = f2bfu(s * u);
      }
    }
  }
  // ---- active[] = 256 * popcount(mask): register counters -> LDS -> global ----
  if (count_active) {
    #pragma unroll
    for (int m = 0; m < 8; ++m)
      atomicAdd(&cnt[arow0 + 16 * m], mcnt[m]);        // ds_add_f32
    __syncthreads();
    for (int i = tid; i < MT; i += 256)
      atomicAdd(&active[t0 + i], 256.0f * cnt[i]);     // global_atomic_add_f32
  }
}

// ---------------------------------------------------------------------------
// Kernel 2: y = h @ down_weight^T + bias. Both tiles double-buffered TDM.
// ---------------------------------------------------------------------------
__global__ __launch_bounds__(256) void cwic_down_kernel(
    const unsigned short* __restrict__ h,     // bf16 [T_TOK][INTER_F]
    const unsigned short* __restrict__ dwb,   // bf16 [OUT_F][INTER_F] (already [n][k])
    const float* __restrict__ db, float* __restrict__ y)
{
  __shared__ __align__(16) unsigned short As[2][MT * TSTR];
  __shared__ __align__(16) unsigned short Bs[2][NT * TSTR];

  const int tid = threadIdx.x;
  const int c0  = blockIdx.x * NT;
  const int t0  = blockIdx.y * MT;
  const int wid = tid >> 5, lane = tid & 31;
  const int hf = lane >> 4, r = lane & 15;
  const int wm = wid & 3,  wn = wid >> 2;

  v8f acc[2][4];
  #pragma unroll
  for (int i = 0; i < 2; ++i)
    #pragma unroll
    for (int j = 0; j < 4; ++j) { v8f z = {}; acc[i][j] = z; }

#if USE_TDM
  if (tid == 0) {
    tdm_load_2d(&h[(size_t)t0 * INTER_F],   (unsigned)(size_t)(void*)As[0], KB, MT, INTER_F);
    tdm_load_2d(&dwb[(size_t)c0 * INTER_F], (unsigned)(size_t)(void*)Bs[0], KB, NT, INTER_F);
  }
#endif

  int p = 0;
  for (int k0 = 0; k0 < INTER_F; k0 += KB, p ^= 1) {
    __syncthreads();
#if USE_TDM
    if (tid == 0) {
      if (k0 + KB < INTER_F) {
        tdm_load_2d(&h[(size_t)t0 * INTER_F + k0 + KB],   (unsigned)(size_t)(void*)As[p ^ 1], KB, MT, INTER_F);
        tdm_load_2d(&dwb[(size_t)c0 * INTER_F + k0 + KB], (unsigned)(size_t)(void*)Bs[p ^ 1], KB, NT, INTER_F);
        __builtin_amdgcn_s_wait_tensorcnt(2);
      } else {
        __builtin_amdgcn_s_wait_tensorcnt(0);
      }
    }
#else
    for (int idx = tid; idx < MT * (KB / 8); idx += 256) {
      int row = idx >> 2, ch = idx & 3;
      *(u32x4*)&As[p][row * TSTR + ch * 8] = *(const u32x4*)&h[(size_t)(t0 + row) * INTER_F + k0 + ch * 8];
      *(u32x4*)&Bs[p][row * TSTR + ch * 8] = *(const u32x4*)&dwb[(size_t)(c0 + row) * INTER_F + k0 + ch * 8];
    }
#endif
    __syncthreads();

    v16bf a[2];
    #pragma unroll
    for (int i = 0; i < 2; ++i)
      a[i] = load_frag16(&As[p][(wm * 32 + i * 16 + r) * TSTR], hf);
    #pragma unroll
    for (int j = 0; j < 4; ++j) {
      v16bf b = load_frag16(&Bs[p][(wn * 64 + j * 16 + r) * TSTR], hf);
      #pragma unroll
      for (int i = 0; i < 2; ++i)
        acc[i][j] = __builtin_amdgcn_wmma_f32_16x16x32_bf16(
            false, a[i], false, b, (short)0, acc[i][j], false, false);
    }
  }

  #pragma unroll
  for (int i = 0; i < 2; ++i) {
    #pragma unroll
    for (int j = 0; j < 4; ++j) {
      int col = c0 + wn * 64 + j * 16 + r;
      float bias = db[col];
      #pragma unroll
      for (int e = 0; e < 8; ++e) {
        int tok = t0 + wm * 32 + i * 16 + e + 8 * hf;
        y[(size_t)tok * OUT_F + col] = acc[i][j][e] + bias;
      }
    }
  }
}

// ---------------------------------------------------------------------------
extern "C" void kernel_launch(void* const* d_in, const int* in_sizes, int n_in,
                              void* d_out, int out_size, void* d_ws, size_t ws_size,
                              hipStream_t stream) {
  const float* x    = (const float*)d_in[0];
  const float* gw   = (const float*)d_in[1];
  const float* gb   = (const float*)d_in[2];
  const float* gthr = (const float*)d_in[3];
  const float* med  = (const float*)d_in[4];
  const float* aad  = (const float*)d_in[5];
  const float* uw   = (const float*)d_in[6];
  const float* ub   = (const float*)d_in[7];
  const float* dw   = (const float*)d_in[8];
  const float* db   = (const float*)d_in[9];

  float* y      = (float*)d_out;
  float* dense  = y + (size_t)T_TOK * OUT_F;
  float* active = dense + T_TOK;

  unsigned short* h_ws = (unsigned short*)d_ws;                    // 32 MB
  unsigned short* gwt  = h_ws + (size_t)T_TOK * INTER_F;           // 32 MB
  unsigned short* uwt  = gwt + (size_t)INTER_F * IN_F;             // 32 MB
  unsigned short* dwb  = uwt + (size_t)INTER_F * IN_F;             // 32 MB
  float* mu_tab  = (float*)(dwb + (size_t)OUT_F * INTER_F);        // 8 KB
  float* thr_tab = mu_tab + IN_F;                                  // 256 KB

  const double pb = pow(0.99, 1000.0);
  const float debiaser  = (float)(1.0 / (1e-7 + (1.0 - pb)));
  const float inv_s2pi  = (float)(1.0 / sqrt(2.0 * M_PI));
  const float thr_scale = (float)sqrt((double)IN_F);

  cwic_init_kernel<<<dim3((T_TOK + 255) / 256), dim3(256), 0, stream>>>(dense, active);
  cwic_tab_kernel<<<dim3((N_STRIPES * IN_F + 255) / 256), dim3(256), 0, stream>>>(
      gthr, med, aad, mu_tab, thr_tab, debiaser, inv_s2pi, thr_scale);

  // per-call weight conditioning: f32 -> bf16, [k][n] -> [n][k]
  cwic_transpose_bf16_kernel<<<dim3(INTER_F / 32, IN_F / 32), dim3(256), 0, stream>>>(
      gw, gwt, IN_F, INTER_F);
  cwic_transpose_bf16_kernel<<<dim3(INTER_F / 32, IN_F / 32), dim3(256), 0, stream>>>(
      uw, uwt, IN_F, INTER_F);
  cwic_convert_bf16_kernel<<<dim3((OUT_F * INTER_F / 2 + 255) / 256), dim3(256), 0, stream>>>(
      dw, dwb, OUT_F * INTER_F / 2);

  cwic_gate_up_kernel<<<dim3(INTER_F / NT, T_TOK / MT), dim3(256), 0, stream>>>(
      x, gwt, uwt, gb, ub, mu_tab, thr_tab, h_ws, active);
  cwic_down_kernel<<<dim3(OUT_F / NT, T_TOK / MT), dim3(256), 0, stream>>>(
      h_ws, dwb, db, y);
}
